// DualGCN_71871982731541
// MI455X (gfx1250) — matrix operand order repaired and checked by reading, not compiled, not run on hardware
//
#include <hip/hip_runtime.h>
#include <stdint.h>

#define DIM 512
#define TILE_M 128
#define TILE_N 128
#define TILE_K 32
#define KSTEPS (DIM / TILE_K)
#define LDS_STRIDE 40            // ushorts per row: 80B pitch -> 16B-aligned frags, conflict padded
#define TILE_SLOT (128 * LDS_STRIDE)

typedef __bf16        v16bf __attribute__((ext_vector_type(16)));
typedef float         v8f   __attribute__((ext_vector_type(8)));
typedef unsigned int  v4u   __attribute__((ext_vector_type(4)));
typedef unsigned int  v2u   __attribute__((ext_vector_type(2)));
typedef int           v4i   __attribute__((ext_vector_type(4)));

#if defined(__has_builtin)
#if __has_builtin(__builtin_amdgcn_global_load_async_to_lds_b128) && \
    __has_builtin(__builtin_amdgcn_s_wait_asynccnt)
#define USE_ASYNC 1
#endif
#endif

#ifdef USE_ASYNC
typedef __attribute__((address_space(1))) v4i* gv4i_ptr;   // global <4 x i32>*
typedef __attribute__((address_space(3))) v4i* lv4i_ptr;   // LDS    <4 x i32>*
#define WAIT_ASYNC(n) __builtin_amdgcn_s_wait_asynccnt(n)
#else
#define WAIT_ASYNC(n) ((void)0)
#endif

union FragAB {
  v16bf v;
  v4u   q[2];
  unsigned int u[8];
};

__device__ __forceinline__ unsigned short f2bf(float f) {
  union { float f; unsigned int u; } v; v.f = f;
  unsigned int u = v.u;
  u += 0x7FFFu + ((u >> 16) & 1u);          // round-to-nearest-even
  return (unsigned short)(u >> 16);
}
__device__ __forceinline__ float bf2f(unsigned short b) {
  union { unsigned int u; float f; } v; v.u = ((unsigned int)b) << 16; return v.f;
}
__device__ __forceinline__ void splitbf(float x, unsigned short& h, unsigned short& l) {
  h = f2bf(x);
  l = f2bf(x - bf2f(h));                    // residual term: ~fp32 accuracy with 3 WMMAs
}

// 16-byte tile-staging copy: async DMA to LDS when available, else load+ds_store.
__device__ __forceinline__ void copy16(const unsigned short* g, unsigned short* l) {
#ifdef USE_ASYNC
  __builtin_amdgcn_global_load_async_to_lds_b128((gv4i_ptr)g, (lv4i_ptr)l, 0, 0);
#else
  *(v4u*)l = *(const v4u*)g;
#endif
}

// ---------------------------------------------------------------------------
// GEMM: C[M x 512] = A[M x 512] @ W[512 x 512] from pre-split bf16 planes.
// AH/AL row-major [M][512]; WTH/WTL transposed [n][k].
// Block tile 128x128, 8 waves, wave = 64 rows x 32 cols = 4x2 WMMA tiles,
// 3 WMMAs per tile (hi*hi + hi*lo + lo*hi). Double-buffered async LDS staging.
// ---------------------------------------------------------------------------
__global__ __launch_bounds__(256)
void k_gemm_planes(const unsigned short* __restrict__ AH, const unsigned short* __restrict__ AL,
                   const unsigned short* __restrict__ WTH, const unsigned short* __restrict__ WTL,
                   float* __restrict__ C, int M)
{
  // planes: 0=XH 1=XL 2=WH 3=WL ; two k-step buffers (80 KB total)
  __shared__ unsigned short lds[2][4][TILE_SLOT];

  const int t    = threadIdx.x;
  const int lane = t & 31;
  const int w    = t >> 5;
  const int half = lane >> 4;      // ISA 16-bit fragment lane-half
  const int r16  = lane & 15;
  const int mq   = w & 1;          // 2 wave-rows  * 64
  const int nq   = w >> 1;         // 4 wave-cols  * 32
  const int blockM = blockIdx.x * TILE_M;
  const int blockN = blockIdx.y * TILE_N;

  v8f acc[4][2];
  const v8f vzero = {0.f,0.f,0.f,0.f,0.f,0.f,0.f,0.f};
#pragma unroll
  for (int i = 0; i < 4; ++i)
#pragma unroll
    for (int j = 0; j < 2; ++j) acc[i][j] = vzero;

  // Stage one k-slice (4 planes x 128 rows x 64B) into buffer b.
  // 8 x 16B chunks per thread -> 8 async instructions per wave.
  auto issue = [&](int b, int k0) {
#pragma unroll
    for (int j = 0; j < 2; ++j) {
      int idx = t + 256 * j;                 // 0..511
      int row = idx >> 2, c = idx & 3;       // row 0..127, 16B chunk 0..3
      int lo  = row * LDS_STRIDE + c * 8;    // ushort offset (16B aligned)
      { // X planes (clamp OOB rows: junk only feeds discarded C rows)
        int gr = blockM + row; if (gr >= M) gr = M - 1;
        size_t go = (size_t)gr * DIM + k0 + c * 8;
        copy16(AH + go, &lds[b][0][lo]);
        copy16(AL + go, &lds[b][1][lo]);
      }
      { // W planes (transposed, row = n)
        size_t go = (size_t)(blockN + row) * DIM + k0 + c * 8;
        copy16(WTH + go, &lds[b][2][lo]);
        copy16(WTL + go, &lds[b][3][lo]);
      }
    }
  };

  auto compute = [&](int b) {
    const unsigned short* XsH = &lds[b][0][0];
    const unsigned short* XsL = &lds[b][1][0];
    const unsigned short* WsH = &lds[b][2][0];
    const unsigned short* WsL = &lds[b][3][0];
    // B fragments (32x16): lane r16 = column; lanes 0-15 hold K 0..15, 16-31 hold K 16..31
    FragAB bh[2], bl[2];
#pragma unroll
    for (int nt = 0; nt < 2; ++nt) {
      int col = nq * 32 + nt * 16 + r16;
      int off = col * LDS_STRIDE + half * 16;
      bh[nt].q[0] = *(const v4u*)&WsH[off];
      bh[nt].q[1] = *(const v4u*)&WsH[off + 8];
      bl[nt].q[0] = *(const v4u*)&WsL[off];
      bl[nt].q[1] = *(const v4u*)&WsL[off + 8];
    }
#pragma unroll
    for (int mt = 0; mt < 4; ++mt) {
      // A fragment (16x32): lane r16 = row; K {half*16..+15}
      FragAB ah, al;
      int row = mq * 64 + mt * 16 + r16;
      int off = row * LDS_STRIDE + half * 16;
      ah.q[0] = *(const v4u*)&XsH[off];
      ah.q[1] = *(const v4u*)&XsH[off + 8];
      al.q[0] = *(const v4u*)&XsL[off];
      al.q[1] = *(const v4u*)&XsL[off + 8];
#pragma unroll
      for (int nt = 0; nt < 2; ++nt) {
        v8f c = acc[mt][nt];
        c = __builtin_amdgcn_wmma_f32_16x16x32_bf16(false, al.v, false, bh[nt].v, (short)0, c, false, false);
        c = __builtin_amdgcn_wmma_f32_16x16x32_bf16(false, ah.v, false, bl[nt].v, (short)0, c, false, false);
        c = __builtin_amdgcn_wmma_f32_16x16x32_bf16(false, ah.v, false, bh[nt].v, (short)0, c, false, false);
        acc[mt][nt] = c;
      }
    }
  };

  issue(0, 0);
#pragma unroll 1
  for (int ks = 0; ks < KSTEPS; ++ks) {
    int cur = ks & 1;
    if (ks + 1 < KSTEPS) {
      issue(cur ^ 1, (ks + 1) * TILE_K);  // prefetch next slice (async)
      WAIT_ASYNC(8);                      // 8 newest (next buf) still in flight; cur buf done
    } else {
      WAIT_ASYNC(0);                      // drain everything
    }
    __syncthreads();                      // all waves' staging visible
    compute(cur);
    __syncthreads();                      // everyone done reading before overwrite
  }

  // C/D layout: VGPR r -> M = r + half*8 (within 16-tile), N = r16
#pragma unroll
  for (int mt = 0; mt < 4; ++mt)
#pragma unroll
    for (int nt = 0; nt < 2; ++nt) {
      int col = blockN + nq * 32 + nt * 16 + r16;
#pragma unroll
      for (int r = 0; r < 8; ++r) {
        int row = blockM + mq * 64 + mt * 16 + half * 8 + r;
        if (row < M) C[(size_t)row * DIM + col] = acc[mt][nt][r];
      }
    }
}

// ---------------------------------------------------------------------------
// fp32 -> bf16 hi/lo plane conversion (row-major)
// ---------------------------------------------------------------------------
__global__ void k_split(const float* __restrict__ A, unsigned short* __restrict__ H,
                        unsigned short* __restrict__ L, int total4) {
  int i = blockIdx.x * blockDim.x + threadIdx.x;  // float4 index
  if (i >= total4) return;
  float4 v = ((const float4*)A)[i];
  unsigned short h0,l0,h1,l1,h2,l2,h3,l3;
  splitbf(v.x,h0,l0); splitbf(v.y,h1,l1); splitbf(v.z,h2,l2); splitbf(v.w,h3,l3);
  v2u th = { ((unsigned)h1 << 16) | h0, ((unsigned)h3 << 16) | h2 };
  v2u tl = { ((unsigned)l1 << 16) | l0, ((unsigned)l3 << 16) | l2 };
  *(v2u*)&H[(size_t)i * 4] = th;
  *(v2u*)&L[(size_t)i * 4] = tl;
}

// W[k][n] fp32 -> transposed bf16 planes WT[n][k]
__global__ void k_splitT(const float* __restrict__ W, unsigned short* __restrict__ H,
                         unsigned short* __restrict__ L) {
  int i = blockIdx.x * blockDim.x + threadIdx.x;
  if (i >= DIM * DIM) return;
  int k = i >> 9, n = i & (DIM - 1);
  unsigned short h, l;
  splitbf(W[i], h, l);
  H[(size_t)n * DIM + k] = h;
  L[(size_t)n * DIM + k] = l;
}

// ---------------------------------------------------------------------------
// Degree / normalization kernels
// ---------------------------------------------------------------------------
__global__ void k_fill1(float* __restrict__ deg, int N) {
  int i = blockIdx.x * blockDim.x + threadIdx.x;
  if (i < N) deg[i] = 1.0f;                 // self-loop contribution
}
__global__ void k_deg_count(const int* __restrict__ dst, float* deg, int E) {
  int i = blockIdx.x * blockDim.x + threadIdx.x;
  int n = gridDim.x * blockDim.x;
  for (int e = i; e < E; e += n) unsafeAtomicAdd(&deg[dst[e]], 1.0f);
}
__global__ void k_rsqrt(float* deg, int N) {
  int i = blockIdx.x * blockDim.x + threadIdx.x;
  if (i < N) deg[i] = rsqrtf(deg[i]);       // deg >= 1 always (self-loops)
}

// out[i,:] = dinv[i]^2 * H[i,:]   (self-loop term; also initializes accumulator)
__global__ void k_selfloop(const float* __restrict__ H, const float* __restrict__ dinv,
                           float* __restrict__ out, int N) {
  int i = blockIdx.x * blockDim.x + threadIdx.x;   // float4 index
  int total = N * (DIM / 4);
  if (i >= total) return;
  int node = i / (DIM / 4);
  float d = dinv[node], s = d * d;
  float4 v = ((const float4*)H)[i];
  v.x *= s; v.y *= s; v.z *= s; v.w *= s;
  ((float4*)out)[i] = v;
}

// wave-per-edge: out[dst,:] += dinv[src]*dinv[dst] * H[src,:]  (L2-resident atomics)
__global__ __launch_bounds__(256)
void k_scatter(const float* __restrict__ H, const int* __restrict__ src,
               const int* __restrict__ dst, const float* __restrict__ dinv,
               float* __restrict__ out, int E)
{
  int lane = threadIdx.x & 31;
  int wid  = (blockIdx.x * blockDim.x + threadIdx.x) >> 5;
  int nw   = (gridDim.x * blockDim.x) >> 5;
  for (int e = wid; e < E; e += nw) {
    int s = src[e], d = dst[e];
    float nrm = dinv[s] * dinv[d];
    const float4* hp = (const float4*)(H + (size_t)s * DIM);
    float* op = out + (size_t)d * DIM;
#pragma unroll
    for (int j = 0; j < 4; ++j) {
      float4 v = hp[lane + 32 * j];
      int c = (lane + 32 * j) * 4;
      unsafeAtomicAdd(op + c + 0, v.x * nrm);
      unsafeAtomicAdd(op + c + 1, v.y * nrm);
      unsafeAtomicAdd(op + c + 2, v.z * nrm);
      unsafeAtomicAdd(op + c + 3, v.w * nrm);
    }
  }
}

// layer-1 epilogue: X1 = relu(a + bias) emitted directly as bf16 hi/lo planes
__global__ void k_bias_relu_split(const float* __restrict__ A, const float* __restrict__ bias,
                                  unsigned short* __restrict__ H, unsigned short* __restrict__ L,
                                  int N) {
  int i = blockIdx.x * blockDim.x + threadIdx.x;   // float4 index
  int total = N * (DIM / 4);
  if (i >= total) return;
  int c4 = i % (DIM / 4);
  float4 v = ((const float4*)A)[i];
  float4 b = ((const float4*)bias)[c4];
  v.x = fmaxf(v.x + b.x, 0.f);
  v.y = fmaxf(v.y + b.y, 0.f);
  v.z = fmaxf(v.z + b.z, 0.f);
  v.w = fmaxf(v.w + b.w, 0.f);
  unsigned short h0,l0,h1,l1,h2,l2,h3,l3;
  splitbf(v.x,h0,l0); splitbf(v.y,h1,l1); splitbf(v.z,h2,l2); splitbf(v.w,h3,l3);
  v2u th = { ((unsigned)h1 << 16) | h0, ((unsigned)h3 << 16) | h2 };
  v2u tl = { ((unsigned)l1 << 16) | l0, ((unsigned)l3 << 16) | l2 };
  *(v2u*)&H[(size_t)i * 4] = th;
  *(v2u*)&L[(size_t)i * 4] = tl;
}

// layer-2 epilogue: y = relu(a + bias) strided into the concat output (fp32)
__global__ void k_bias_relu(const float* __restrict__ A, const float* __restrict__ bias,
                            float* __restrict__ out, int ldOut, int colOff, int N) {
  int i = blockIdx.x * blockDim.x + threadIdx.x;   // float4 index
  int total = N * (DIM / 4);
  if (i >= total) return;
  int node = i / (DIM / 4);
  int c4   = i % (DIM / 4);
  float4 v = ((const float4*)A)[i];
  float4 b = ((const float4*)bias)[c4];
  v.x = fmaxf(v.x + b.x, 0.f);
  v.y = fmaxf(v.y + b.y, 0.f);
  v.z = fmaxf(v.z + b.z, 0.f);
  v.w = fmaxf(v.w + b.w, 0.f);
  ((float4*)(out + (size_t)node * ldOut + colOff))[c4] = v;
}

// ---------------------------------------------------------------------------
extern "C" void kernel_launch(void* const* d_in, const int* in_sizes, int n_in,
                              void* d_out, int out_size, void* d_ws, size_t ws_size,
                              hipStream_t stream)
{
  const float* x   = (const float*)d_in[0];
  const int*   ei  = (const int*)  d_in[1];
  const int*   eic = (const int*)  d_in[2];
  const float* W1  = (const float*)d_in[3];
  const float* b1  = (const float*)d_in[4];
  const float* W2  = (const float*)d_in[5];
  const float* b2  = (const float*)d_in[6];
  const float* Wc1 = (const float*)d_in[7];
  const float* bc1 = (const float*)d_in[8];
  const float* Wc2 = (const float*)d_in[9];
  const float* bc2 = (const float*)d_in[10];

  const int N  = in_sizes[0] / DIM;   // 50000
  const int E1 = in_sizes[1] / 2;     // 1.6M
  const int E2 = in_sizes[2] / 2;

  // workspace carve-up (256B aligned)
  char*  base = (char*)d_ws;
  size_t off  = 0;
  auto alloc = [&](size_t bytes) {
    char* p = base + off;
    off = (off + bytes + 255) & ~(size_t)255;
    return p;
  };
  float*          dinv   = (float*)         alloc((size_t)N * 4);
  unsigned short* planeH = (unsigned short*)alloc((size_t)N * DIM * 2);
  unsigned short* planeL = (unsigned short*)alloc((size_t)N * DIM * 2);
  unsigned short* wtH    = (unsigned short*)alloc((size_t)DIM * DIM * 2);
  unsigned short* wtL    = (unsigned short*)alloc((size_t)DIM * DIM * 2);
  float*          bufH   = (float*)         alloc((size_t)N * DIM * 4);
  float*          bufA   = (float*)         alloc((size_t)N * DIM * 4);

  const dim3 gemmGrid((N + TILE_M - 1) / TILE_M, DIM / TILE_N);
  const int  ewGrid = (N * (DIM / 4) + 255) / 256;
  const int  nGrid  = (N + 255) / 256;
  const int  wGrid  = (DIM * DIM + 255) / 256;
  float* outF = (float*)d_out;

  auto branch = [&](const int* e, int E, const float* Wa, const float* ba,
                    const float* Wb, const float* bb, int colOff) {
    const int* srcp = e;
    const int* dstp = e + E;
    // symmetric normalization terms (deg includes self-loop)
    k_fill1    <<<nGrid, 256, 0, stream>>>(dinv, N);
    k_deg_count<<<2048, 256, 0, stream>>>(dstp, dinv, E);
    k_rsqrt    <<<nGrid, 256, 0, stream>>>(dinv, N);
    // layer 1
    k_split    <<<ewGrid, 256, 0, stream>>>(x, planeH, planeL, N * (DIM / 4));
    k_splitT   <<<wGrid, 256, 0, stream>>>(Wa, wtH, wtL);
    k_gemm_planes<<<gemmGrid, 256, 0, stream>>>(planeH, planeL, wtH, wtL, bufH, N);
    k_selfloop <<<ewGrid, 256, 0, stream>>>(bufH, dinv, bufA, N);
    k_scatter  <<<16384, 256, 0, stream>>>(bufH, srcp, dstp, dinv, bufA, E);
    k_bias_relu_split<<<ewGrid, 256, 0, stream>>>(bufA, ba, planeH, planeL, N);
    // layer 2
    k_splitT   <<<wGrid, 256, 0, stream>>>(Wb, wtH, wtL);
    k_gemm_planes<<<gemmGrid, 256, 0, stream>>>(planeH, planeL, wtH, wtL, bufH, N);
    k_selfloop <<<ewGrid, 256, 0, stream>>>(bufH, dinv, bufA, N);
    k_scatter  <<<16384, 256, 0, stream>>>(bufH, srcp, dstp, dinv, bufA, E);
    k_bias_relu<<<ewGrid, 256, 0, stream>>>(bufA, bb, outF, 2 * DIM, colOff, N);
  };

  branch(ei,  E1, W1,  b1,  W2,  b2,  0);
  branch(eic, E2, Wc1, bc1, Wc2, bc2, DIM);
}
